// WaveletAttentionFusion_20675972563166
// MI455X (gfx1250) — matrix operand order repaired and checked by reading, not compiled, not run on hardware
//
#include <hip/hip_runtime.h>
#include <hip/hip_bf16.h>

// Fused WaveletAttentionFusion for MI455X (gfx1250, wave32, WMMA).
// permute+bf16 -> weight precvt -> QKV proj (WMMA) ->
// flash attention (2-way kv split per tile, cooperative double-buffered
// async K/V staging to LDS, WMMA S/PV, l via P*ones WMMA, row-max via
// xor-shuffle, LDS C->A bounce, in-block split merge) -> final proj (WMMA).

#define NTOK   8192      // H*W tokens
#define CDIM   128
#define HWSZ   8192      // 64*128
#define HBUF   1048576   // elements per bf16 scratch buffer (2 MB)

#if defined(__AMDGCN__) && __has_builtin(__builtin_amdgcn_global_load_async_to_lds_b128)
#define USE_ASYNC_LDS 1
#else
#define USE_ASYNC_LDS 0
#endif

typedef __attribute__((ext_vector_type(16))) __bf16 v16bf;
typedef __attribute__((ext_vector_type(8)))  float  v8f;
typedef int v4i_vs __attribute__((vector_size(4 * sizeof(int))));

union Frag16 { v16bf v; uint4 q[2]; unsigned short h[16]; };

static __device__ __forceinline__ unsigned short f2bf(float f) {
  __bf16 h = (__bf16)f;
  return __builtin_bit_cast(unsigned short, h);
}
static __device__ __forceinline__ unsigned pack2bf(float a, float b) {
  return (unsigned)f2bf(a) | ((unsigned)f2bf(b) << 16);
}
static __device__ __forceinline__ uint4 pack8_bf(const float* v) {
  uint4 o;
  o.x = pack2bf(v[0], v[1]);
  o.y = pack2bf(v[2], v[3]);
  o.z = pack2bf(v[4], v[5]);
  o.w = pack2bf(v[6], v[7]);
  return o;
}

// A fragment 16x32 (M=m0.., K=k0..) from row-major bf16 [rows][ld].
static __device__ __forceinline__ v16bf load_a_frag(const unsigned short* base,
                                                    int ld, int m0, int k0, int lane) {
  int m  = m0 + (lane & 15);
  int g8 = (lane & 16) ? 8 : 0;
  const unsigned short* p = base + (long)m * ld + k0;
  Frag16 f;
  f.q[0] = *(const uint4*)(p + g8);
  f.q[1] = *(const uint4*)(p + 16 + g8);
  return f.v;
}

// B fragment 32x16 where element (k,n) = base[(n0+n)*ld + k0 + k].
static __device__ __forceinline__ v16bf load_b_fragT(const unsigned short* base,
                                                     int ld, int k0, int n0, int lane) {
  int n  = n0 + (lane & 15);
  int kg = (lane & 16) ? 8 : 0;
  const unsigned short* p = base + (long)n * ld + k0 + kg;
  Frag16 f;
  f.q[0] = *(const uint4*)(p);
  f.q[1] = *(const uint4*)(p + 16);
  return f.v;
}

static __device__ __forceinline__ v8f wmma_bf16(v16bf a, v16bf b, v8f c) {
  return __builtin_amdgcn_wmma_f32_16x16x32_bf16(false, a, false, b, (short)0, c,
                                                 false, false);
}

static __device__ __forceinline__ v8f vzero8() {
  v8f z;
#pragma unroll
  for (int r = 0; r < 8; ++r) z[r] = 0.0f;
  return z;
}

static __device__ __forceinline__ v16bf ones_frag() {
  Frag16 f;
#pragma unroll
  for (int j = 0; j < 16; ++j) f.h[j] = 0x3F80;   // bf16 1.0
  return f.v;
}

// 16-byte global -> LDS copy: async DMA path on CDNA5, reg bounce fallback.
static __device__ __forceinline__ void stage16(const unsigned short* g,
                                               unsigned short* l) {
#if USE_ASYNC_LDS
  __builtin_amdgcn_global_load_async_to_lds_b128(
      (__attribute__((address_space(1))) v4i_vs*)(unsigned long long)(const void*)g,
      (__attribute__((address_space(3))) v4i_vs*)l, 0, 0);
#else
  *(uint4*)l = *(const uint4*)g;
#endif
}

static __device__ __forceinline__ void wait_async_all() {
#if USE_ASYNC_LDS
#if __has_builtin(__builtin_amdgcn_s_wait_asynccnt)
  __builtin_amdgcn_s_wait_asynccnt(0);
#else
  asm volatile("s_wait_asynccnt 0" ::: "memory");
#endif
#endif
}

// ---------------------------------------------------------------------------
// Stage 0a: window-permute x / wavelet into bf16 [N][C] row-major (pair-packed).
// ---------------------------------------------------------------------------
__global__ __launch_bounds__(256) void wa_permute(const float* __restrict__ x,
                                                  const float* __restrict__ wf,
                                                  unsigned* __restrict__ xw2,
                                                  unsigned* __restrict__ ww2) {
  int i = blockIdx.x * 256 + threadIdx.x;          // 0 .. 2*(N*C/2)-1
  int which = (i >= NTOK * CDIM / 2);
  int jp = which ? i - NTOK * CDIM / 2 : i;        // pair index
  int t = jp >> 6, c = (jp & 63) * 2;
  int win = t >> 6, rr = t & 63;
  int wh = rr >> 3, ww = rr & 7;
  int hb = win >> 4, wb = win & 15;
  int sp = (hb * 8 + wh) * 128 + wb * 8 + ww;      // h_*W + w_
  const float* src = which ? wf : x;
  float v0 = src[(long)c * HWSZ + sp];
  float v1 = src[(long)(c + 1) * HWSZ + sp];
  (which ? ww2 : xw2)[jp] = pack2bf(v0, v1);
}

// Stage 0b: convert Wq,Wk,Wv,Wp (f32 [128][128]) -> bf16 row-major, pair-packed.
__global__ __launch_bounds__(256) void wa_wcvt(const float* __restrict__ Wq,
                                               const float* __restrict__ Wk,
                                               const float* __restrict__ Wv,
                                               const float* __restrict__ Wp,
                                               unsigned* __restrict__ wbf2) {
  int i = blockIdx.x * 256 + threadIdx.x;          // 0 .. 4*8192-1 (pairs)
  int mat = i >> 13, p = i & 8191;
  const float* W = (mat == 0) ? Wq : (mat == 1) ? Wk : (mat == 2) ? Wv : Wp;
  wbf2[i] = pack2bf(W[p * 2], W[p * 2 + 1]);
}

// ---------------------------------------------------------------------------
// Stage 1: QKV projection. One wave = one 16x16 output tile.
// ---------------------------------------------------------------------------
__global__ __launch_bounds__(256) void wa_qkv(const unsigned short* __restrict__ xw,
                                              const unsigned short* __restrict__ wwb,
                                              const unsigned short* __restrict__ wbf,
                                              const float* __restrict__ bq,
                                              const float* __restrict__ bk,
                                              const float* __restrict__ bv,
                                              unsigned short* __restrict__ qo,
                                              unsigned short* __restrict__ ko,
                                              unsigned short* __restrict__ vT) {
  int lane = threadIdx.x & 31;
  int wid  = blockIdx.x * 8 + (threadIdx.x >> 5);  // 0..12287
  int mat  = wid / 4096;                           // 0=q 1=k 2=v
  int tile = wid - mat * 4096;
  int t0 = (tile >> 3) * 16, n0 = (tile & 7) * 16;

  const unsigned short* src = (mat == 0) ? xw : wwb;
  const unsigned short* W   = wbf + (long)mat * CDIM * CDIM;
  const float* bias = (mat == 0) ? bq : (mat == 1 ? bk : bv);

  v8f acc = vzero8();
#pragma unroll
  for (int k0 = 0; k0 < 128; k0 += 32) {
    v16bf a = load_a_frag(src, CDIM, t0, k0, lane);
    v16bf b = load_b_fragT(W, CDIM, k0, n0, lane);   // (k,n) = W[n][k]
    acc = wmma_bf16(a, b, acc);
  }

  int n  = n0 + (lane & 15);
  int hi = (lane >> 4) & 1;                        // C layout: rows r+8*hi
  float bz = bias[n];

  if (mat == 2) {                                  // vT[n][t0+8*hi..+7] contiguous
    float vals[8];
#pragma unroll
    for (int r = 0; r < 8; ++r) vals[r] = acc[r] + bz;
    *(uint4*)(vT + (long)n * NTOK + t0 + 8 * hi) = pack8_bf(vals);
  } else {
    float s = (mat == 0) ? 0.125f : 1.0f;          // fold softmax scale into q
    unsigned short* dst = (mat == 0) ? qo : ko;
#pragma unroll
    for (int r = 0; r < 8; ++r) {
      int t = t0 + r + 8 * hi;
      dst[(long)t * CDIM + n] = f2bf((acc[r] + bz) * s);
    }
  }
}

// ---------------------------------------------------------------------------
// Stage 2: flash attention. Block = 4 q-tiles x 2 kv-splits (one head).
// Each 4-wave split-group cooperatively stages its 32-token K/V tile into LDS
// (double buffered, async DMA), then all waves consume via ds_load fragments.
// Per iter/wave: 4 WMMA (S) + row-max shuffle + 1 WMMA (l=P*ones) + 4 WMMA (PV).
// ---------------------------------------------------------------------------
__global__ __launch_bounds__(256) void wa_attn(const unsigned short* __restrict__ qb,
                                               const unsigned short* __restrict__ kb,
                                               const unsigned short* __restrict__ vT,
                                               unsigned short* __restrict__ ob) {
  __shared__ __align__(16) unsigned short kstage[2][2][32 * 64]; // [split][buf] 16KB
  __shared__ __align__(16) unsigned short vstage[2][2][64 * 32]; // [split][buf] 16KB
  __shared__ __align__(16) unsigned short plds[8][16 * 32];      // P staging 8KB
  __shared__ float xchg[4][32][49];                              // merge ~25KB

  int lane  = threadIdx.x & 31;
  int wave  = threadIdx.x >> 5;
  int split = wave & 1;                            // kv half
  int lt    = wave >> 1;                           // local tile 0..3
  int tile  = blockIdx.x * 4 + lt;                 // 0..1023 (same head per block)
  int head  = tile >> 9;
  int qt    = tile & 511;
  int t0    = qt * 16;
  int h64   = head * 64;
  int kv0   = split * (NTOK / 2);
  int gl    = lt * 32 + lane;                      // lane id within split group
  unsigned short* pl = plds[wave];

  v16bf qa0 = load_a_frag(qb, CDIM, t0, h64 + 0,  lane);
  v16bf qa1 = load_a_frag(qb, CDIM, t0, h64 + 32, lane);
  v16bf onesb = ones_frag();

  v8f acc[4];
#pragma unroll
  for (int f = 0; f < 4; ++f) acc[f] = vzero8();
  v8f l_acc = vzero8();
  float m_r[8];
#pragma unroll
  for (int r = 0; r < 8; ++r) m_r[r] = -3.0e38f;

  // cooperative stage of K[32][64] + V[64][32] for one 32-token step
  auto stage_kv = [&](int j0, int buf) {
    unsigned short* kl = kstage[split][buf];
    unsigned short* vl = vstage[split][buf];
#pragma unroll
    for (int s = 0; s < 2; ++s) {                  // K: 256 chunks / 128 lanes
      int q = gl + s * 128;
      int n = q >> 3, c8 = (q & 7) * 8;
      stage16(kb + (long)(j0 + n) * CDIM + h64 + c8, kl + n * 64 + c8);
    }
#pragma unroll
    for (int s = 0; s < 2; ++s) {                  // V: 256 chunks / 128 lanes
      int q = gl + s * 128;
      int c = q >> 2, k8 = (q & 3) * 8;
      stage16(vT + (long)(h64 + c) * NTOK + j0 + k8, vl + c * 32 + k8);
    }
  };

  const int NIT = (NTOK / 2) / 32;                 // 128
  stage_kv(kv0, 0);

  for (int it = 0; it < NIT; ++it) {
    int buf = it & 1;
    wait_async_all();
    __syncthreads();                               // staged tile visible to all
    if (it + 1 < NIT) stage_kv(kv0 + (it + 1) * 32, buf ^ 1);  // prefetch next

    const unsigned short* kl = kstage[split][buf];
    const unsigned short* vl = vstage[split][buf];

    v16bf k00 = load_b_fragT(kl, 64, 0,  0,  lane);
    v16bf k01 = load_b_fragT(kl, 64, 32, 0,  lane);
    v16bf k10 = load_b_fragT(kl, 64, 0,  16, lane);
    v16bf k11 = load_b_fragT(kl, 64, 32, 16, lane);
    v8f s0 = vzero8(), s1 = vzero8();
    s0 = wmma_bf16(qa0, k00, s0);
    s0 = wmma_bf16(qa1, k01, s0);
    s1 = wmma_bf16(qa0, k10, s1);
    s1 = wmma_bf16(qa1, k11, s1);

    float rescale[8];
#pragma unroll
    for (int r = 0; r < 8; ++r) {
      float a = s0[r], b = s1[r];
      float mx = fmaxf(a, b);
#pragma unroll
      for (int mk = 1; mk <= 8; mk <<= 1) mx = fmaxf(mx, __shfl_xor(mx, mk));
      float mn = fmaxf(m_r[r], mx);
      float sc = __expf(m_r[r] - mn);
      float p0 = __expf(a - mn);
      float p1 = __expf(b - mn);
      m_r[r] = mn;
      rescale[r] = sc;
      int M = r + ((lane & 16) ? 8 : 0);           // C-layout -> row-major LDS
      pl[M * 32 + (lane & 15)]      = f2bf(p0);
      pl[M * 32 + 16 + (lane & 15)] = f2bf(p1);
    }

    v16bf pa;                                      // P back as A fragment
    {
      int m = lane & 15, g8 = (lane & 16) ? 8 : 0;
      Frag16 f;
      f.q[0] = *(const uint4*)(pl + m * 32 + g8);
      f.q[1] = *(const uint4*)(pl + m * 32 + 16 + g8);
      pa = f.v;
    }

    {                                              // l += rowsum(P) via WMMA
      v8f t;
#pragma unroll
      for (int r = 0; r < 8; ++r) t[r] = l_acc[r] * rescale[r];
      l_acc = wmma_bf16(pa, onesb, t);
    }

#pragma unroll
    for (int f = 0; f < 4; ++f) {                  // O = O*rescale + P*V
      v16bf vf = load_b_fragT(vl, 32, 0, f * 16, lane);
      v8f t;
#pragma unroll
      for (int r = 0; r < 8; ++r) t[r] = acc[f][r] * rescale[r];
      acc[f] = wmma_bf16(pa, vf, t);
    }
  }

  // ---- merge the two kv-splits of this tile through LDS ----
  int pair = lt;
  float* xb = &xchg[pair][lane][0];
  if (split == 1) {
#pragma unroll
    for (int f = 0; f < 4; ++f)
#pragma unroll
      for (int r = 0; r < 8; ++r) xb[f * 8 + r] = acc[f][r];
#pragma unroll
    for (int r = 0; r < 8; ++r) { xb[32 + r] = m_r[r]; xb[40 + r] = l_acc[r]; }
  }
  __syncthreads();
  if (split == 0) {
    float eA[8], eB[8], linv[8];
#pragma unroll
    for (int r = 0; r < 8; ++r) {
      float mB = xb[32 + r], lB = xb[40 + r];
      float mn = fmaxf(m_r[r], mB);
      eA[r] = __expf(m_r[r] - mn);
      eB[r] = __expf(mB - mn);
      linv[r] = 1.0f / (l_acc[r] * eA[r] + lB * eB[r]);
    }
    int hi = (lane >> 4) & 1;                      // write un-permuted [C][H][W]
    int tb = t0 + 8 * hi;
    int win = tb >> 6, rr = tb & 63;
    int wh = rr >> 3;
    int hb = win >> 4, wb = win & 15;
    int sp = (hb * 8 + wh) * 128 + wb * 8;
#pragma unroll
    for (int f = 0; f < 4; ++f) {
      int c = h64 + f * 16 + (lane & 15);
      float vals[8];
#pragma unroll
      for (int r = 0; r < 8; ++r)
        vals[r] = (acc[f][r] * eA[r] + xb[f * 8 + r] * eB[r]) * linv[r];
      *(uint4*)(ob + (long)c * HWSZ + sp) = pack8_bf(vals);
    }
  }
}

// ---------------------------------------------------------------------------
// Stage 3: final projection along W axis: y[c,h,:] = o[c,h,:] @ Wp^T + bp.
// ---------------------------------------------------------------------------
__global__ __launch_bounds__(256) void wa_proj(const unsigned short* __restrict__ ob,
                                               const unsigned short* __restrict__ wpbf,
                                               const float* __restrict__ bp,
                                               float* __restrict__ out) {
  int lane = threadIdx.x & 31;
  int wid  = blockIdx.x * 8 + (threadIdx.x >> 5);  // 0..4095
  int t0 = (wid >> 3) * 16, n0 = (wid & 7) * 16;

  v8f acc = vzero8();
#pragma unroll
  for (int k0 = 0; k0 < 128; k0 += 32) {
    v16bf a = load_a_frag(ob, CDIM, t0, k0, lane);
    v16bf b = load_b_fragT(wpbf, CDIM, k0, n0, lane);
    acc = wmma_bf16(a, b, acc);
  }

  int n  = n0 + (lane & 15);
  int hi = (lane >> 4) & 1;
  float bz = bp[n];
#pragma unroll
  for (int r = 0; r < 8; ++r) {
    int row = t0 + r + 8 * hi;                     // row = c*64 + h_
    int c = row >> 6, h_ = row & 63;
    out[(long)c * HWSZ + h_ * 128 + n] = acc[r] + bz;
  }
}

// ---------------------------------------------------------------------------
extern "C" void kernel_launch(void* const* d_in, const int* in_sizes, int n_in,
                              void* d_out, int out_size, void* d_ws, size_t ws_size,
                              hipStream_t stream) {
  const float* x  = (const float*)d_in[0];
  const float* wf = (const float*)d_in[1];
  const float* Wq = (const float*)d_in[2];
  const float* bq = (const float*)d_in[3];
  const float* Wk = (const float*)d_in[4];
  const float* bk = (const float*)d_in[5];
  const float* Wv = (const float*)d_in[6];
  const float* bv = (const float*)d_in[7];
  const float* Wp = (const float*)d_in[8];
  const float* bp = (const float*)d_in[9];
  float* out = (float*)d_out;

  unsigned short* xw  = (unsigned short*)d_ws;     // 6 x 2MB bf16 + 128KB weights
  unsigned short* wwb = xw  + HBUF;
  unsigned short* qb  = wwb + HBUF;
  unsigned short* kb  = qb  + HBUF;
  unsigned short* vT  = kb  + HBUF;
  unsigned short* ob  = vT  + HBUF;
  unsigned short* wbf = ob  + HBUF;                // [4][128][128] bf16
  unsigned short* wpbf = wbf + 3 * CDIM * CDIM;

  wa_permute<<<4096, 256, 0, stream>>>(x, wf, (unsigned*)xw, (unsigned*)wwb);
  wa_wcvt<<<128, 256, 0, stream>>>(Wq, Wk, Wv, Wp, (unsigned*)wbf);
  wa_qkv<<<1536, 256, 0, stream>>>(xw, wwb, wbf, bq, bk, bv, qb, kb, vT);
  wa_attn<<<256, 256, 0, stream>>>(qb, kb, vT, ob);
  wa_proj<<<512, 256, 0, stream>>>(ob, wpbf, bp, out);
}
